// GeographicPruning_48782238548325
// MI455X (gfx1250) — compile-verified, matrix operand order checked by reading, not run on hardware
//
#include <hip/hip_runtime.h>
#include <hip/hip_bf16.h>
#include <math.h>

// Problem constants (fixed shapes from the reference)
#define B_    2
#define N_    20000
#define D_    128
#define L_    400
#define P_    200      // NUM_POS
#define K_    1024     // GEO_K
#define EPS_  1e-8f
#define SIGMA_ 50.0f
#define HALFW_ 5.0f

typedef __attribute__((ext_vector_type(16))) _Float16     v16h;
typedef __attribute__((ext_vector_type(8)))  float        v8f;
typedef __attribute__((ext_vector_type(4)))  float        f32x4;
typedef __attribute__((ext_vector_type(4)))  unsigned int v4u;
typedef __attribute__((ext_vector_type(8)))  int          v8i;
typedef __attribute__((ext_vector_type(4)))  int          v4i;

__device__ __forceinline__ unsigned orderKey(float f) {
  unsigned u = __float_as_uint(f);
  return (u & 0x80000000u) ? ~u : (u | 0x80000000u);   // monotonic: bigger float -> bigger key
}
__device__ __forceinline__ float keyToFloat(unsigned k) {
  unsigned u = (k & 0x80000000u) ? (k ^ 0x80000000u) : ~k;
  return __uint_as_float(u);
}

// ---------------------------------------------------------------------------
// Kernel 1: normalized LUT columns. One wave32 per (b,l).
// Each lane computes 7 erf-difference values per table (224 slots, p<200 valid).
// Normalization sum runs on the matrix unit: A = data (7 halves/lane of the
// 16x32 f16 A tile), B = ones  =>  D[i][j] = rowsum_i (f32).  Per-lane sum of
// the 8 D components gives S_lo (lanes 0-15) / S_hi (lanes 16-31); shfl_xor(16)
// combines them into the exact total of all packed slots.
// Output layout: wsl[row][0..199] = lut_x column, wsl[row][200..399] = lut_y
// column (contiguous 1600 B per row -> single TDM tile for phase 2).
// ---------------------------------------------------------------------------
__global__ __launch_bounds__(32)
void geo_lut_wmma_kernel(const float* __restrict__ latent,
                         const float* __restrict__ centers,
                         float* __restrict__ wsl) {
  const int row  = blockIdx.x;      // b*L_ + l
  const int lane = threadIdx.x;     // wave32
  const float mux = latent[row * 2 + 0];
  const float muy = latent[row * 2 + 1];
  const float inv_s = 1.0f / (SIGMA_ * 1.41421356237f);

  float vx[7], vy[7];
#pragma unroll
  for (int t = 0; t < 7; ++t) {
    int p = t * 32 + lane;
    float c = (p < P_) ? centers[p] : 0.0f;
    float ex_hi = erff((c + HALFW_ - mux) * inv_s);
    float ex_lo = erff((c - HALFW_ - mux) * inv_s);
    float ey_hi = erff((c + HALFW_ - muy) * inv_s);
    float ey_lo = erff((c - HALFW_ - muy) * inv_s);
    vx[t] = (p < P_) ? 0.5f * (ex_hi - ex_lo) : 0.0f;
    vy[t] = (p < P_) ? 0.5f * (ey_hi - ey_lo) : 0.0f;
  }

  v16h ax, ay, ones;
#pragma unroll
  for (int i = 0; i < 16; ++i) {
    ax[i] = (_Float16)0.0f; ay[i] = (_Float16)0.0f; ones[i] = (_Float16)1.0f;
  }
#pragma unroll
  for (int t = 0; t < 7; ++t) { ax[t] = (_Float16)vx[t]; ay[t] = (_Float16)vy[t]; }

  v8f accx = {}; v8f accy = {};
  // D = A x Ones + 0  -> rowsums (f32 accumulate), EXEC is all-ones here.
  accx = __builtin_amdgcn_wmma_f32_16x16x32_f16(false, ax, false, ones, (short)0,
                                                accx, false, false);
  accy = __builtin_amdgcn_wmma_f32_16x16x32_f16(false, ay, false, ones, (short)0,
                                                accy, false, false);

  float sx = 0.0f, sy = 0.0f;
#pragma unroll
  for (int i = 0; i < 8; ++i) { sx += accx[i]; sy += accy[i]; }
  sx += __shfl_xor(sx, 16, 32);   // S_lo + S_hi == sum over all 224 packed slots
  sy += __shfl_xor(sy, 16, 32);

  const float invx = 1.0f / (sx + EPS_);
  const float invy = 1.0f / (sy + EPS_);
#pragma unroll
  for (int t = 0; t < 7; ++t) {
    int p = t * 32 + lane;
    if (p < P_) {
      wsl[(size_t)row * (2 * P_) + p]      = vx[t] * invx;
      wsl[(size_t)row * (2 * P_) + P_ + p] = vy[t] * invy;
    }
  }
}

// ---------------------------------------------------------------------------
// Kernel 1b: extract per-token (x_idx, y_idx) once into a packed u32 array so
// phase-2 only streams a 160 KB L2-resident array instead of strided token rows.
// ---------------------------------------------------------------------------
__global__ __launch_bounds__(256)
void geo_packidx_kernel(const float* __restrict__ tokens,
                        unsigned* __restrict__ idxpack) {
  int i = blockIdx.x * 256 + threadIdx.x;
  if (i >= B_ * N_) return;
  int xi = (int)tokens[(size_t)i * D_ + 1];
  int yi = (int)tokens[(size_t)i * D_ + 2];
  xi = min(max(xi, 0), P_ - 1);
  yi = min(max(yi, 0), P_ - 1);
  idxpack[i] = (unsigned)xi | ((unsigned)yi << 16);
}

// ---------------------------------------------------------------------------
// Kernel 2: per (b,l) row -> scores -> top-1024 (radix select + bitonic sort).
// Dynamic LDS layout (90880 B total; 320 KB/WGP => 3 resident blocks):
//   [0,80000)        u32 keys[20000]
//   [80000,84096)    u32 sel_key[1024]
//   [84096,88192)    s32 sel_idx[1024]
//   [88192,88992)    f32 lutx[200]   \ contiguous 1600 B tile,
//   [88992,89792)    f32 luty[200]   / filled by one TENSOR_LOAD_TO_LDS
//   [89792,90816)    u32 histo[256]
//   [90816,90880)    u32 ctrl[16]  (0:prefix 1:remaining 2:total_gt 3:cnt_gt 4:cnt_eq)
// ---------------------------------------------------------------------------
#define SMEM_BYTES_ 90880

__global__ __launch_bounds__(512)
void geo_topk_kernel(const unsigned* __restrict__ idxpack,
                     const float* __restrict__ wsl,
                     float* __restrict__ bias_out,
                     int* __restrict__ idx_out) {
  extern __shared__ unsigned char smem[];
  unsigned* keys_s  = (unsigned*)(smem);
  unsigned* sel_key = (unsigned*)(smem + 80000);
  int*      sel_idx = (int*)     (smem + 84096);
  float*    lutx_s  = (float*)   (smem + 88192);
  float*    luty_s  = (float*)   (smem + 88992);
  unsigned* histo   = (unsigned*)(smem + 89792);
  unsigned* ctrl    = (unsigned*)(smem + 90816);

  const int row = blockIdx.x;          // b*L_ + l
  const int b   = row / L_;
  const int tid = threadIdx.x;
  const int nthr = blockDim.x;

  // ---- Stage the 1600 B LUT tile into LDS via the Tensor Data Mover --------
#if __has_builtin(__builtin_amdgcn_tensor_load_to_lds)
  if (tid < 32) {                       // wave 0 issues; TENSORcnt is per-wave
    unsigned long long g = (unsigned long long)(uintptr_t)(wsl + (size_t)row * (2 * P_));
    unsigned lds_off = (unsigned)(uintptr_t)lutx_s;   // low 32 bits = LDS byte addr
    // D# group 0: count=1 | lds_addr | global_addr[56:0] | type=2 ("image")
    v4u g0 = { 1u,
               lds_off,
               (unsigned)g,
               (unsigned)((g >> 32) & 0x01FFFFFFull) | (2u << 30) };
    // D# group 1: data_size=2 (4 B), tensor_dim0=400, tensor_dim1=1,
    //             tile_dim0=400, tile_dim1=1, tensor_dim0_stride=400
    v8i g1 = { (int)(2u << 16),         // [17:16] data_size = 4 bytes
               (int)(400u << 16),       // [79:48] tensor_dim0 = 400 (low half)
               (int)(1u << 16),         // tensor_dim0 hi = 0 | tensor_dim1 = 1
               (int)(400u << 16),       // tensor_dim1 hi = 0 | [127:112] tile_dim0 = 400
               (int)(1u),               // [143:128] tile_dim1 = 1, tile_dim2 = 0
               (int)(400u),             // [207:160] tensor_dim0_stride = 400
               0, 0 };
    v4i gz4 = { 0, 0, 0, 0 };           // groups 2/3 unused (<=2-D tensor)
    v8i gz8 = { 0, 0, 0, 0, 0, 0, 0, 0 };
    // 6-arg form on this toolchain (clang expected 6 args)
    __builtin_amdgcn_tensor_load_to_lds(g0, g1, gz4, gz4, gz8, 0);
#if __has_builtin(__builtin_amdgcn_s_wait_tensorcnt)
    __builtin_amdgcn_s_wait_tensorcnt(0);
#else
    asm volatile("s_wait_tensorcnt 0x0" ::: "memory");
#endif
  }
#else
  for (int q = tid; q < 2 * P_; q += nthr)          // fallback: plain loads
    lutx_s[q] = wsl[(size_t)row * (2 * P_) + q];    // luty_s == lutx_s + P_
#endif
  if (tid == 0) { ctrl[0] = 0u; ctrl[1] = K_; ctrl[2] = 0u; ctrl[3] = 0u; ctrl[4] = 0u; }
  __syncthreads();

  // score -> ordered key
  for (int n = tid; n < N_; n += nthr) {
    __builtin_prefetch(&idxpack[(size_t)b * N_ + n + 4 * nthr], 0, 0);
    unsigned pk = idxpack[(size_t)b * N_ + n];
    int xi = (int)(pk & 0xFFFFu);
    int yi = (int)(pk >> 16);
    float g  = lutx_s[xi] * luty_s[yi];
    float sc = logf(g + EPS_);
    keys_s[n] = orderKey(sc);
  }
  __syncthreads();

  // 4-pass MSB-first radix select of the K_-th largest key
  for (int pass = 0; pass < 4; ++pass) {
    const int shift = 24 - 8 * pass;
    for (int i = tid; i < 256; i += nthr) histo[i] = 0u;
    __syncthreads();
    const unsigned prefix = ctrl[0];
    for (int n = tid; n < N_; n += nthr) {
      unsigned key = keys_s[n];
      if (pass == 0 || (key >> (shift + 8)) == prefix)
        atomicAdd(&histo[(key >> shift) & 0xFFu], 1u);
    }
    __syncthreads();
    if (tid == 0) {
      unsigned rem = ctrl[1];
      unsigned cum = 0u;
      int chosen = 0;
      for (int bb = 255; bb >= 0; --bb) {
        unsigned c = histo[bb];
        if (cum + c >= rem) { chosen = bb; break; }
        cum += c;
      }
      ctrl[0] = (prefix << 8) | (unsigned)chosen;
      ctrl[1] = rem - cum;        // still needed among keys == new prefix
      ctrl[2] += cum;             // # keys strictly greater so far
    }
    __syncthreads();
  }

  const unsigned T        = ctrl[0];
  const unsigned need_eq  = ctrl[1];
  const unsigned total_gt = ctrl[2];

  // compaction: all keys > T, plus need_eq keys == T
  for (int n = tid; n < N_; n += nthr) {
    unsigned key = keys_s[n];
    if (key > T) {
      unsigned s0 = atomicAdd(&ctrl[3], 1u);
      sel_key[s0] = key; sel_idx[s0] = n;
    } else if (key == T) {
      unsigned e = atomicAdd(&ctrl[4], 1u);
      if (e < need_eq) { sel_key[total_gt + e] = key; sel_idx[total_gt + e] = n; }
    }
  }
  __syncthreads();

  // bitonic sort (ascending) of 1024 (key, idx) pairs
  for (unsigned sz = 2; sz <= K_; sz <<= 1) {
    for (unsigned st = sz >> 1; st > 0; st >>= 1) {
      for (unsigned i = tid; i < K_; i += nthr) {
        unsigned j = i ^ st;
        if (j > i) {
          unsigned ki = sel_key[i], kj = sel_key[j];
          bool up = ((i & sz) == 0);
          if ((ki > kj) == up) {
            sel_key[i] = kj; sel_key[j] = ki;
            int t = sel_idx[i]; sel_idx[i] = sel_idx[j]; sel_idx[j] = t;
          }
        }
      }
      __syncthreads();
    }
  }

  // emit descending (bias is streamed out non-temporally; idx stays hot in L2
  // because the gather kernel consumes it immediately)
  for (int j = tid; j < K_; j += nthr) {
    int srt = K_ - 1 - j;
    __builtin_nontemporal_store(keyToFloat(sel_key[srt]),
                                &bias_out[(size_t)row * K_ + j]);
    idx_out[(size_t)row * K_ + j] = sel_idx[srt];
  }
}

// ---------------------------------------------------------------------------
// Kernel 3: gather. One wave32 per output row (512 B). tokens is 20.5 MB and
// stays L2-resident; the 419 MB streaming output uses non-temporal stores so
// it does not evict the gather working set. (A TDM gather-mode variant would
// route this through LDS twice — strictly worse than L2->VGPR->NT-store.)
// ---------------------------------------------------------------------------
__global__ __launch_bounds__(256)
void geo_gather_kernel(const float* __restrict__ tokens,
                       const int* __restrict__ mask,
                       const int* __restrict__ idxsel,
                       float* __restrict__ out_toks,
                       float* __restrict__ out_msks) {
  const int wave = threadIdx.x >> 5;
  const int lane = threadIdx.x & 31;
  const size_t r = (size_t)blockIdx.x * 8 + wave;     // < B_*L_*K_
  const int n = idxsel[r];
  const int b = (int)(r / ((size_t)L_ * K_));

  const f32x4* src = (const f32x4*)(tokens + ((size_t)b * N_ + (size_t)n) * D_);
  f32x4 v = src[lane];                                 // 32 lanes x 16 B = 512 B row
  f32x4* dst = (f32x4*)(out_toks + r * D_);
  __builtin_nontemporal_store(v, dst + lane);
  if (lane == 0) {
    float m = mask[(size_t)b * N_ + (size_t)n] ? 1.0f : 0.0f;
    __builtin_nontemporal_store(m, out_msks + r);
  }
}

// ---------------------------------------------------------------------------
extern "C" void kernel_launch(void* const* d_in, const int* in_sizes, int n_in,
                              void* d_out, int out_size, void* d_ws, size_t ws_size,
                              hipStream_t stream) {
  const float* tokens  = (const float*)d_in[0];        // B,N,D f32
  const int*   mask    = (const int*)  d_in[1];        // B,N   i32
  const float* latent  = (const float*)d_in[2];        // B,L,2 f32
  const float* centers = (const float*)d_in[3];        // P     f32

  float* out = (float*)d_out;
  float* out_toks = out;                                          // B*L*K*D
  float* out_msks = out_toks + (size_t)B_ * L_ * K_ * D_;         // B*L*K
  float* out_bias = out_msks + (size_t)B_ * L_ * K_;              // B*L*K

  // workspace carve (≈4.7 MB)
  float*    wsl     = (float*)d_ws;                               // B*L * 2P (x|y)
  unsigned* idxpack = (unsigned*)(wsl + (size_t)B_ * L_ * 2 * P_);// B*N
  int*      idxsel  = (int*)(idxpack + (size_t)B_ * N_);          // B*L*K

  // K1: LUT columns with WMMA normalization reduction
  geo_lut_wmma_kernel<<<B_ * L_, 32, 0, stream>>>(latent, centers, wsl);

  // K1b: pack token x/y indices
  geo_packidx_kernel<<<(B_ * N_ + 255) / 256, 256, 0, stream>>>(tokens, idxpack);

  // K2: per-row score + top-K (select + sort), TDM-staged LUT, writes bias+idx
  geo_topk_kernel<<<B_ * L_, 512, SMEM_BYTES_, stream>>>(idxpack, wsl,
                                                         out_bias, idxsel);

  // K3: gather token rows + masks
  geo_gather_kernel<<<(B_ * L_ * K_) / 8, 256, 0, stream>>>(tokens, mask, idxsel,
                                                            out_toks, out_msks);
}
